// AttentionSingleHead_9929964388794
// MI455X (gfx1250) — matrix operand order repaired and checked by reading, not compiled, not run on hardware
//
#include <hip/hip_runtime.h>
#include <hip/hip_bf16.h>

typedef _Float16 v16h __attribute__((ext_vector_type(16)));
typedef _Float16 v8h  __attribute__((ext_vector_type(8)));
typedef float    v8f  __attribute__((ext_vector_type(8)));
typedef float    f32x4 __attribute__((ext_vector_type(4)));
typedef unsigned int u32x4 __attribute__((ext_vector_type(4)));
typedef int i32x4 __attribute__((ext_vector_type(4)));
typedef int i32x8 __attribute__((ext_vector_type(8)));

union Frag {
    v16h h;
    u32x4 u[2];
};

static constexpr int E    = 1024;
static constexpr int H    = 128;
static constexpr int Tseq = 4096;
static constexpr int Bb   = 4;
static constexpr int BT   = Bb * Tseq; // 16384

// -------------------------------------------------------------------------
// Kernel 0: bulk fp32 -> f16 conversion (8 elements / thread, b128 in/out)
// -------------------------------------------------------------------------
__global__ __launch_bounds__(256, 1)
void cvt_kernel(const float* __restrict__ src, _Float16* __restrict__ dst, int n8)
{
    const int idx = blockIdx.x * 256 + threadIdx.x;
    if (idx >= n8) return;
    const f32x4* s = (const f32x4*)src + (size_t)idx * 2;
    f32x4 a = s[0], b = s[1];
    v8h o;
    #pragma unroll
    for (int j = 0; j < 4; ++j) { o[j] = (_Float16)a[j]; o[4 + j] = (_Float16)b[j]; }
    *(v8h*)(dst + (size_t)idx * 8) = o;
}

// -------------------------------------------------------------------------
// TDM descriptor build + issue (6-arg builtin).
// 2D tile copy: tile rows of dim0 8-byte elements, dim1 rows, row stride
// stride0 (8-byte units), packed row-major into LDS at lds_off.
// -------------------------------------------------------------------------
__device__ __forceinline__ void tdm_load_2d(unsigned lds_off, const void* gaddr,
                                            unsigned dim0, unsigned dim1,
                                            unsigned stride0,
                                            unsigned tile0, unsigned tile1)
{
    const unsigned long long ga = (unsigned long long)gaddr;
    u32x4 g0;
    g0[0] = 1u;                                    // count=1 (valid), user mode
    g0[1] = lds_off;                               // lds_addr
    g0[2] = (unsigned)(ga & 0xffffffffu);          // global_addr[31:0]
    g0[3] = (unsigned)((ga >> 32) & 0x01ffffffu)   // global_addr[56:32]
          | (2u << 30);                            // type=2 (image)
    i32x8 g1;
    g1[0] = (int)(3u << 16);                                   // data_size=8B
    g1[1] = (int)((dim0 & 0xffffu) << 16);                     // tensor_dim0[15:0]
    g1[2] = (int)(((dim0 >> 16) & 0xffffu) | ((dim1 & 0xffffu) << 16));
    g1[3] = (int)(((dim1 >> 16) & 0xffffu) | ((tile0 & 0xffffu) << 16));
    g1[4] = (int)(tile1 & 0xffffu);                            // tile_dim1, tile_dim2=0
    g1[5] = (int)stride0;                                      // tensor_dim0_stride[31:0]
    g1[6] = 0;
    g1[7] = 0;
    i32x4 z4 = {0, 0, 0, 0};
    i32x8 z8 = {0, 0, 0, 0, 0, 0, 0, 0};
    __builtin_amdgcn_tensor_load_to_lds(g0, g1, z4, z4, z8, 0);
}

__device__ __forceinline__ unsigned lds_offset_of(const void* p)
{
    return (unsigned)(unsigned long long)p; // LDS flat addr: low 32 bits = LDS offset
}

// -------------------------------------------------------------------------
// Kernel 1: QKV projection from pre-converted f16. y = x @ W^T, W:[H,E].
// Each wave: 32(M) x 128(N) strip (two A frags share every B frag -> 2 WMMAs
// per B load), K=1024 in 32-chunks. Q,K stored f16 [BT][H]; V transposed [H][BT].
// -------------------------------------------------------------------------
__global__ __launch_bounds__(128, 1)
void qkv_kernel(const _Float16* __restrict__ xh,
                const _Float16* __restrict__ wh,   // [3][H][E] f16
                _Float16* __restrict__ qh,
                _Float16* __restrict__ kh,
                _Float16* __restrict__ vT)
{
    const int tid  = threadIdx.x;
    const int wave = tid >> 5;
    const int lane = tid & 31;
    const int lr   = lane & 15;
    const int g    = lane >> 4;
    const int row0 = blockIdx.x * 128 + wave * 32; // 32 rows per wave

    #pragma unroll
    for (int mat = 0; mat < 3; ++mat) {
        const _Float16* W = wh + (size_t)mat * H * E;
        v8f acc0[8] = {};
        v8f acc1[8] = {};

        for (int kc = 0; kc < E / 32; ++kc) {
            // Two A fragments (rows row0..+15 and row0+16..+31)
            Frag a0, a1;
            const _Float16* xr0 = xh + (size_t)(row0 + lr) * E + kc * 32 + (g ? 16 : 0);
            const _Float16* xr1 = xr0 + (size_t)16 * E;
            a0.u[0] = *(const u32x4*)(xr0);
            a0.u[1] = *(const u32x4*)(xr0 + 8);
            a1.u[0] = *(const u32x4*)(xr1);
            a1.u[1] = *(const u32x4*)(xr1 + 8);
            #pragma unroll
            for (int n = 0; n < 8; ++n) {
                // B fragment 32x16: B[k][n] = W[n][k], 16 contiguous halfs
                const _Float16* wr = W + (size_t)(n * 16 + lr) * E + kc * 32 + (g ? 16 : 0);
                v16h bh = *(const v16h*)wr;
                acc0[n] = __builtin_amdgcn_wmma_f32_16x16x32_f16(
                    false, a0.h, false, bh, (short)0, acc0[n], false, false);
                acc1[n] = __builtin_amdgcn_wmma_f32_16x16x32_f16(
                    false, a1.h, false, bh, (short)0, acc1[n], false, false);
            }
        }

        // C/D layout: VGPR i, lane L -> M = i + (L/16)*8, N = L%16
        if (mat < 2) {
            _Float16* dst = (mat == 0) ? qh : kh;
            #pragma unroll
            for (int n = 0; n < 8; ++n) {
                #pragma unroll
                for (int i = 0; i < 8; ++i) {
                    dst[(size_t)(row0 + g * 8 + i) * H + n * 16 + lr]      = (_Float16)acc0[n][i];
                    dst[(size_t)(row0 + 16 + g * 8 + i) * H + n * 16 + lr] = (_Float16)acc1[n][i];
                }
            }
        } else {
            #pragma unroll
            for (int n = 0; n < 8; ++n) {
                v8h p0, p1;
                #pragma unroll
                for (int i = 0; i < 8; ++i) { p0[i] = (_Float16)acc0[n][i]; p1[i] = (_Float16)acc1[n][i]; }
                _Float16* vrow = vT + (size_t)(n * 16 + lr) * BT + row0 + g * 8;
                *(v8h*)(vrow)      = p0;
                *(v8h*)(vrow + 16) = p1;
            }
        }
    }
}

// -------------------------------------------------------------------------
// Kernel 2: causal flash attention, TDM double-buffered K/V tiles in LDS.
// -------------------------------------------------------------------------
__device__ __forceinline__ float redmax16(float v) {
    #pragma unroll
    for (int m = 8; m; m >>= 1) v = fmaxf(v, __shfl_xor(v, m, 16));
    return v;
}
__device__ __forceinline__ float redsum16(float v) {
    #pragma unroll
    for (int m = 8; m; m >>= 1) v += __shfl_xor(v, m, 16);
    return v;
}

__global__ __launch_bounds__(128, 1)
void attn_kernel(const _Float16* __restrict__ qh,
                 const _Float16* __restrict__ kh,
                 const _Float16* __restrict__ vT,
                 float* __restrict__ out)
{
    __shared__ __align__(32) _Float16 kTile[2][64][128];   // [buf][key][h]
    __shared__ __align__(32) _Float16 vTile[2][128][64];   // [buf][h][key]
    __shared__ __align__(32) _Float16 pTile[4][16][64];    // per-wave P

    const int tid  = threadIdx.x;
    const int wave = tid >> 5;
    const int lane = tid & 31;
    const int lr   = lane & 15;
    const int g    = lane >> 4;
    const int b    = blockIdx.y;
    const int qbase = blockIdx.x * 64;
    const int rowQ  = b * Tseq + qbase + wave * 16;

    // Preload Q fragments (4 h-chunks of 16x32)
    Frag qf[4];
    #pragma unroll
    for (int hc = 0; hc < 4; ++hc) {
        const _Float16* src = qh + (size_t)(rowQ + lr) * H + hc * 32 + (g ? 16 : 0);
        qf[hc].u[0] = *(const u32x4*)(src);
        qf[hc].u[1] = *(const u32x4*)(src + 8);
    }

    v8f accO[8] = {};
    float mrow[8], lrow[8];
    #pragma unroll
    for (int i = 0; i < 8; ++i) { mrow[i] = -1e30f; lrow[i] = 0.0f; }

    const float smScale = 0.08838834764831845f; // 1/sqrt(128)
    const int nTiles = blockIdx.x + 1;

    // Prologue: wave 0 issues TDM for tile 0 into buffer 0.
    if (wave == 0) {
        // K tile: 64 rows x 128 halfs contiguous = 2D 32x64 of 8B elems
        tdm_load_2d(lds_offset_of(&kTile[0][0][0]),
                    kh + (size_t)(b * Tseq) * H, 32, 64, 32, 32, 64);
        // V^T tile: 128 rows x 16 8B-elems, row stride BT/4 8B-elems
        tdm_load_2d(lds_offset_of(&vTile[0][0][0]),
                    vT + (size_t)(b * Tseq), 16, 128, BT / 4, 16, 128);
    }

    int cur = 0;
    for (int jt = 0; jt < nTiles; ++jt) {
        const int keyBase = jt * 64;

        if (wave == 0) __builtin_amdgcn_s_wait_tensorcnt(0); // tile jt landed
        __syncthreads();                                      // visible to all waves
                                                              // (also fences prev reads)
        // Issue next tile into the other buffer while computing on this one.
        if (jt + 1 < nTiles && wave == 0) {
            const int nb = keyBase + 64;
            tdm_load_2d(lds_offset_of(&kTile[cur ^ 1][0][0]),
                        kh + (size_t)(b * Tseq + nb) * H, 32, 64, 32, 32, 64);
            tdm_load_2d(lds_offset_of(&vTile[cur ^ 1][0][0]),
                        vT + (size_t)(b * Tseq + nb), 16, 128, BT / 4, 16, 128);
        }

        // S = Q K^T : 4 key-subtiles x 4 h-chunks
        v8f s[4] = {};
        #pragma unroll
        for (int t = 0; t < 4; ++t) {
            #pragma unroll
            for (int hc = 0; hc < 4; ++hc) {
                const _Float16* src = &kTile[cur][t * 16 + lr][hc * 32 + (g ? 16 : 0)];
                v16h bh = *(const v16h*)src;
                s[t] = __builtin_amdgcn_wmma_f32_16x16x32_f16(
                    false, qf[hc].h, false, bh, (short)0, s[t], false, false);
            }
        }

        // Scale + causal mask (element (q = i+g*8, key = t*16+lr))
        #pragma unroll
        for (int t = 0; t < 4; ++t) {
            const int key = keyBase + t * 16 + lr;
            #pragma unroll
            for (int i = 0; i < 8; ++i) {
                const int q = qbase + wave * 16 + i + g * 8;
                float v = s[t][i] * smScale;
                s[t][i] = (key <= q) ? v : -1e30f;
            }
        }

        // Online softmax (row = i + g*8, values across 16 lanes)
        float mnew[8], rscale[8];
        #pragma unroll
        for (int i = 0; i < 8; ++i) {
            float mt = fmaxf(fmaxf(s[0][i], s[1][i]), fmaxf(s[2][i], s[3][i]));
            mt = redmax16(mt);
            mnew[i] = fmaxf(mrow[i], mt);
            rscale[i] = __expf(mrow[i] - mnew[i]);
            mrow[i] = mnew[i];
        }
        #pragma unroll
        for (int t = 0; t < 4; ++t) {
            #pragma unroll
            for (int i = 0; i < 8; ++i) {
                float p = __expf(s[t][i] - mnew[i]);
                s[t][i] = p;
                pTile[wave][i + g * 8][t * 16 + lr] = (_Float16)p;
            }
        }
        #pragma unroll
        for (int i = 0; i < 8; ++i) {
            float rs = redsum16(s[0][i] + s[1][i] + s[2][i] + s[3][i]);
            lrow[i] = lrow[i] * rscale[i] + rs;
        }
        #pragma unroll
        for (int n = 0; n < 8; ++n)
            #pragma unroll
            for (int i = 0; i < 8; ++i) accO[n][i] *= rscale[i];

        // O += P V : 2 key-chunks of 32 x 8 h-tiles
        #pragma unroll
        for (int kc = 0; kc < 2; ++kc) {
            Frag pa;
            const _Float16* psrc = &pTile[wave][lr][kc * 32 + (g ? 16 : 0)];
            pa.u[0] = *(const u32x4*)(psrc);
            pa.u[1] = *(const u32x4*)(psrc + 8);
            #pragma unroll
            for (int n = 0; n < 8; ++n) {
                const _Float16* vsrc = &vTile[cur][n * 16 + lr][kc * 32 + (g ? 16 : 0)];
                v16h bh = *(const v16h*)vsrc;
                accO[n] = __builtin_amdgcn_wmma_f32_16x16x32_f16(
                    false, pa.h, false, bh, (short)0, accO[n], false, false);
            }
        }

        cur ^= 1; // next iteration's top barrier orders buffer reuse
    }

    // Normalize, write fp32 [B,T,H]
    #pragma unroll
    for (int n = 0; n < 8; ++n)
        #pragma unroll
        for (int i = 0; i < 8; ++i)
            out[(size_t)(rowQ + i + g * 8) * H + n * 16 + lr] = accO[n][i] / lrow[i];
}

// -------------------------------------------------------------------------
extern "C" void kernel_launch(void* const* d_in, const int* in_sizes, int n_in,
                              void* d_out, int out_size, void* d_ws, size_t ws_size,
                              hipStream_t stream)
{
    const float* x  = (const float*)d_in[0];
    const float* Wq = (const float*)d_in[1];
    const float* Wk = (const float*)d_in[2];
    const float* Wv = (const float*)d_in[3];

    _Float16* qh = (_Float16*)d_ws;                 //  4 MB
    _Float16* kh = qh + (size_t)BT * H;             //  4 MB
    _Float16* vT = kh + (size_t)BT * H;             //  4 MB
    _Float16* xh = vT + (size_t)BT * H;             // 32 MB
    _Float16* wh = xh + (size_t)BT * E;             // 0.75 MB
    float* out = (float*)d_out;

    // fp32 -> f16 pre-pass
    const int n8x = BT * E / 8;
    cvt_kernel<<<dim3((n8x + 255) / 256), dim3(256), 0, stream>>>(x, xh, n8x);
    const int n8w = H * E / 8;
    cvt_kernel<<<dim3((n8w + 255) / 256), dim3(256), 0, stream>>>(Wq, wh, n8w);
    cvt_kernel<<<dim3((n8w + 255) / 256), dim3(256), 0, stream>>>(Wk, wh + (size_t)H * E, n8w);
    cvt_kernel<<<dim3((n8w + 255) / 256), dim3(256), 0, stream>>>(Wv, wh + (size_t)2 * H * E, n8w);

    qkv_kernel<<<dim3(BT / 128), dim3(128), 0, stream>>>(xh, wh, qh, kh, vT);
    attn_kernel<<<dim3(Tseq / 64, Bb), dim3(128), 0, stream>>>(qh, kh, vT, out);
}
